// LSTMModel_67697274520016
// MI455X (gfx1250) — matrix operand order repaired and checked by reading, not compiled, not run on hardware
//
#include <hip/hip_runtime.h>

#define DIMS   2048
#define VOCABN 512
#define TSTEPS 8192
#define NB_SCAN 256   // persistent workgroups for the scan

typedef __attribute__((ext_vector_type(16))) __bf16 v16bf;
typedef __attribute__((ext_vector_type(8)))  float  v8f;
typedef __attribute__((ext_vector_type(4)))  unsigned int v4u;
typedef __attribute__((ext_vector_type(8)))  int v8i_;
typedef __attribute__((ext_vector_type(4)))  int v4i;

union FragBF { unsigned int u[8]; v16bf v; };

__device__ __forceinline__ unsigned short f2bf(float x) {
    union { float f; unsigned int u; } c; c.f = x;
    unsigned int u = c.u;
    unsigned int r = (u + 0x7FFFu + ((u >> 16) & 1u)) >> 16;
    return (unsigned short)r;
}
__device__ __forceinline__ float bf2f(unsigned int h) {
    union { unsigned int u; float f; } c; c.u = h << 16;
    return c.f;
}

// ---- Tensor DMA descriptor builders (D#, 2D tile of 2-byte elements) ----
// group0: [1:0]=count=1, [63:32]=lds_addr, [120:64]=global_addr, [127:126]=type=2
__device__ __forceinline__ v4u tdm_g0(unsigned lds_addr, unsigned long long gaddr) {
    v4u g;
    g.x = 1u;
    g.y = lds_addr;
    g.z = (unsigned)(gaddr & 0xFFFFFFFFull);
    g.w = (unsigned)((gaddr >> 32) & 0x01FFFFFFull) | (2u << 30);
    return g;
}
// group1: [17:16]=data_size(1->2B), [79:48]=tensor_dim0, [111:80]=tensor_dim1,
//         [127:112]=tile_dim0, [143:128]=tile_dim1, [207:160]=tensor_dim0_stride
__device__ __forceinline__ v8i_ tdm_g1(unsigned td0, unsigned td1,
                                       unsigned tile0, unsigned tile1,
                                       unsigned long long stride0) {
    v8i_ g;
    g[0] = (int)(1u << 16);
    g[1] = (int)((td0 & 0xFFFFu) << 16);
    g[2] = (int)(((td0 >> 16) & 0xFFFFu) | ((td1 & 0xFFFFu) << 16));
    g[3] = (int)(((td1 >> 16) & 0xFFFFu) | ((tile0 & 0xFFFFu) << 16));
    g[4] = (int)(tile1 & 0xFFFFu);                 // tile_dim2 = 0
    g[5] = (int)(stride0 & 0xFFFFFFFFull);
    g[6] = (int)((stride0 >> 32) & 0xFFFFull);     // dim1_stride unused (2D)
    g[7] = 0;
    return g;
}

// ---------------------------------------------------------------- utilities
__global__ __launch_bounds__(256) void cvt_f32_to_bf16(const float* __restrict__ src,
                                                       unsigned short* __restrict__ dst,
                                                       long long n) {
    long long i = (long long)blockIdx.x * 256 + threadIdx.x;
    long long stride = (long long)gridDim.x * 256;
    for (; i < n; i += stride) dst[i] = f2bf(src[i]);
}

// Wx[2048][512] -> WxT[512][2048]
__global__ __launch_bounds__(256) void transpose_wx(const float* __restrict__ Wx,
                                                    float* __restrict__ WxT) {
    int i = blockIdx.x * 256 + threadIdx.x;      // over 512*2048
    int v = i >> 11, d = i & (DIMS - 1);
    WxT[i] = Wx[(size_t)d * VOCABN + v];
}

// x_bf[t][d] = bf16(WxT[tok[t]][d] + bx[d])
__global__ __launch_bounds__(256) void embed_kernel(const int* __restrict__ tokens,
                                                    const float* __restrict__ WxT,
                                                    const float* __restrict__ bx,
                                                    unsigned short* __restrict__ x_bf) {
    long long i = (long long)blockIdx.x * 256 + threadIdx.x;   // over T*DIMS
    int t = (int)(i >> 11), d = (int)(i & (DIMS - 1));
    int tok = tokens[t];
    x_bf[i] = f2bf(WxT[(size_t)tok * DIMS + d] + bx[d]);
}

// ------------------------------------------------- WMMA bf16 GEMM: C = A*B^T + bias
// A: [M][K] bf16 row-major, B: [N][K] bf16 row-major, C: [M][N] f32.
// Block tile 128x64, 8 waves of 32x32 (2x2 WMMA 16x16x32).
// K staged TK=32 at a time via Tensor Data Mover into double-buffered LDS.
#define TM 128
#define TN 64
#define TK 32
__global__ __launch_bounds__(256)
void wmma_gemm_bf16(const unsigned short* __restrict__ A,
                    const unsigned short* __restrict__ B,
                    const float* __restrict__ bias,
                    float* __restrict__ C,
                    int M, int N, int K) {
    __shared__ unsigned short As[2][TM * TK];   // 2 x 8 KB
    __shared__ unsigned short Bs[2][TN * TK];   // 2 x 4 KB

    const int tid  = threadIdx.x;
    const int wave = tid >> 5;
    const int lane = tid & 31;
    const int r    = lane & 15;
    const int hi   = lane >> 4;
    const int wm   = wave >> 1;          // 0..3  (M direction)
    const int wn   = wave & 1;           // 0..1  (N direction)
    const int m0   = blockIdx.y * TM;
    const int n0   = blockIdx.x * TN;

    const unsigned long long Abase = (unsigned long long)A + ((size_t)m0 * K) * 2ull;
    const unsigned long long Bbase = (unsigned long long)B + ((size_t)n0 * K) * 2ull;
    const v4i z4 = {0, 0, 0, 0};
    const v8i_ z8 = {0, 0, 0, 0, 0, 0, 0, 0};
    const int nchunk = K / TK;

    // issue TDM loads for chunk c into buffer buf (wave 0 only)
    auto issue_chunk = [&](int c, int buf) {
        unsigned long long koff = (unsigned long long)(c * TK) * 2ull;
        v4u a0 = tdm_g0((unsigned)(unsigned long long)&As[buf][0], Abase + koff);
        v8i_ a1 = tdm_g1((unsigned)K, (unsigned)M, TK, TM, (unsigned long long)K);
        __builtin_amdgcn_tensor_load_to_lds(a0, a1, z4, z4, z8, 0);
        v4u b0 = tdm_g0((unsigned)(unsigned long long)&Bs[buf][0], Bbase + koff);
        v8i_ b1 = tdm_g1((unsigned)K, (unsigned)N, TK, TN, (unsigned long long)K);
        __builtin_amdgcn_tensor_load_to_lds(b0, b1, z4, z4, z8, 0);
    };

    v8f acc[2][2] = {};

    if (wave == 0) issue_chunk(0, 0);

    for (int c = 0; c < nchunk; ++c) {
        const int buf = c & 1;
        if (wave == 0) {
            if (c + 1 < nchunk) {
                issue_chunk(c + 1, buf ^ 1);
                __builtin_amdgcn_s_wait_tensorcnt(2);  // chunk c has landed
            } else {
                __builtin_amdgcn_s_wait_tensorcnt(0);
            }
        }
        __syncthreads();   // publish LDS tile to all waves

        FragBF a[2], b[2];
        #pragma unroll
        for (int s = 0; s < 2; ++s) {
            int mr = wm * 32 + s * 16 + r;
            const unsigned int* Au = (const unsigned int*)&As[buf][0] + mr * (TK / 2);
            #pragma unroll
            for (int p = 0; p < 8; ++p)
                a[s].u[p] = Au[p + 4 * (hi + (p >= 4 ? 1 : 0))];
        }
        #pragma unroll
        for (int s = 0; s < 2; ++s) {
            int nr = wn * 32 + s * 16 + r;
            const unsigned int* Bu = (const unsigned int*)&Bs[buf][0] + nr * (TK / 2);
            #pragma unroll
            for (int p = 0; p < 8; ++p)
                b[s].u[p] = Bu[p + 4 * (hi + (p >= 4 ? 1 : 0))];
        }
        #pragma unroll
        for (int i = 0; i < 2; ++i)
            #pragma unroll
            for (int j = 0; j < 2; ++j)
                acc[i][j] = __builtin_amdgcn_wmma_f32_16x16x32_bf16(
                    false, a[i].v, false, b[j].v, (short)0, acc[i][j], false, false);

        __syncthreads();   // all waves done reading buf before TDM reuses it
    }

    #pragma unroll
    for (int i = 0; i < 2; ++i)
        #pragma unroll
        for (int j = 0; j < 2; ++j) {
            int col = n0 + wn * 32 + j * 16 + r;
            float bv = bias[col];
            #pragma unroll
            for (int e = 0; e < 8; ++e) {
                int rowm = m0 + wm * 32 + i * 16 + e + 8 * hi;
                C[(size_t)rowm * N + col] = acc[i][j][e] + bv;
            }
        }
}

// ------------------------------------------------- persistent LSTM scan
// U4bf: [4][2048][2048] bf16 (f,i,c,o). gates: [4][T][2048] f32 (x-side preacts incl bw bias).
// preact: double buffered [2][8192] f32. h/c state replicated in LDS per WG.
__global__ __launch_bounds__(256)
void lstm_scan(const unsigned short* __restrict__ U4bf,
               const float* __restrict__ buf_, const float* __restrict__ bui,
               const float* __restrict__ buc,  const float* __restrict__ buo,
               const float* __restrict__ gates,
               float* __restrict__ preact,
               int* __restrict__ bar_cnt,
               unsigned short* __restrict__ hs_bf) {
    __shared__ float hsh[DIMS];
    __shared__ float csh[DIMS];

    const int tid  = threadIdx.x;
    const int wave = tid >> 5;
    const int lane = tid & 31;

    for (int j = tid; j < DIMS; j += 256) { hsh[j] = 0.f; csh[j] = 0.f; }
    __syncthreads();

    int bar_gen = 0;

    for (int t = 0; t < TSTEPS; ++t) {
        float* pa = preact + (size_t)(t & 1) * 8192;

        // ---- phase A: this WG computes 32 of the 8192 recurrent dot products
        #pragma unroll
        for (int ov = 0; ov < 4; ++ov) {
            int o = blockIdx.x * 32 + wave * 4 + ov;          // 0..8191
            const unsigned int* Uu = (const unsigned int*)U4bf + ((size_t)o << 10);
            if (ov < 3)  // warm the next row of U into cache off the critical path
                __builtin_prefetch((const void*)(Uu + 1024 + lane * 32), 0, 1);
            const float2* h2 = (const float2*)hsh;
            float accv = 0.f;
            #pragma unroll 8
            for (int i = 0; i < 32; ++i) {
                unsigned int u = Uu[i * 32 + lane];
                float2 h = h2[i * 32 + lane];
                accv += bf2f(u & 0xFFFFu) * h.x + bf2f(u >> 16) * h.y;
            }
            #pragma unroll
            for (int off = 16; off > 0; off >>= 1)
                accv += __shfl_xor(accv, off, 32);
            if (lane == 0) pa[o] = accv;
        }

        // ---- device-wide barrier (release preact, acquire others')
        __syncthreads();
        if (tid == 0) {
            __threadfence();
            ++bar_gen;
            __hip_atomic_fetch_add(bar_cnt, 1, __ATOMIC_ACQ_REL, __HIP_MEMORY_SCOPE_AGENT);
            int target = bar_gen * NB_SCAN;
            while (__hip_atomic_load(bar_cnt, __ATOMIC_ACQUIRE, __HIP_MEMORY_SCOPE_AGENT) < target)
                __builtin_amdgcn_s_sleep(2);
        }
        __syncthreads();

        // ---- phase B: every WG redundantly updates its replicated h/c state
        const float* gbase = gates + (size_t)t * DIMS;
        for (int j = tid; j < DIMS; j += 256) {
            float pf = pa[j]          + buf_[j] + gbase[(size_t)0 * TSTEPS * DIMS + j];
            float pi = pa[2048 + j]   + bui[j]  + gbase[(size_t)1 * TSTEPS * DIMS + j];
            float pc = pa[4096 + j]   + buc[j]  + gbase[(size_t)2 * TSTEPS * DIMS + j];
            float po = pa[6144 + j]   + buo[j]  + gbase[(size_t)3 * TSTEPS * DIMS + j];
            float fg = 1.f / (1.f + __expf(-pf));
            float ig = 1.f / (1.f + __expf(-pi));
            float cn = tanhf(pc);
            float og = 1.f / (1.f + __expf(-po));
            float c  = fg * csh[j] + ig * cn;
            float h  = og * tanhf(c);
            csh[j] = c;
            hsh[j] = h;
            if (blockIdx.x == 0) hs_bf[(size_t)t * DIMS + j] = f2bf(h);
        }
        __syncthreads();
    }
}

// ------------------------------------------------- in-place row log_softmax (N=512)
__global__ __launch_bounds__(256) void log_softmax_rows(float* __restrict__ y, int N) {
    __shared__ float red[256];
    float* row = y + (size_t)blockIdx.x * N;
    int tid = threadIdx.x;

    float m = -3.4e38f;
    for (int j = tid; j < N; j += 256) m = fmaxf(m, row[j]);
    red[tid] = m; __syncthreads();
    for (int s = 128; s > 0; s >>= 1) { if (tid < s) red[tid] = fmaxf(red[tid], red[tid + s]); __syncthreads(); }
    m = red[0]; __syncthreads();

    float ssum = 0.f;
    for (int j = tid; j < N; j += 256) ssum += __expf(row[j] - m);
    red[tid] = ssum; __syncthreads();
    for (int s = 128; s > 0; s >>= 1) { if (tid < s) red[tid] += red[tid + s]; __syncthreads(); }
    float lse = m + __logf(red[0]); __syncthreads();

    for (int j = tid; j < N; j += 256) row[j] = row[j] - lse;
}

// ---------------------------------------------------------------- launcher
extern "C" void kernel_launch(void* const* d_in, const int* in_sizes, int n_in,
                              void* d_out, int out_size, void* d_ws, size_t ws_size,
                              hipStream_t stream) {
    (void)in_sizes; (void)n_in; (void)out_size; (void)ws_size;

    const int*   tokens = (const int*)  d_in[0];
    const float* Wx  = (const float*)d_in[1];
    const float* bx  = (const float*)d_in[2];
    const float* Uf  = (const float*)d_in[3];
    const float* buf_= (const float*)d_in[4];
    const float* Wf  = (const float*)d_in[5];
    const float* bwf = (const float*)d_in[6];
    const float* Ui  = (const float*)d_in[7];
    const float* bui = (const float*)d_in[8];
    const float* Wi  = (const float*)d_in[9];
    const float* bwi = (const float*)d_in[10];
    const float* Uc  = (const float*)d_in[11];
    const float* buc = (const float*)d_in[12];
    const float* Wc  = (const float*)d_in[13];
    const float* bwc = (const float*)d_in[14];
    const float* Uo  = (const float*)d_in[15];
    const float* buo = (const float*)d_in[16];
    const float* Wo  = (const float*)d_in[17];
    const float* bwo = (const float*)d_in[18];
    const float* V   = (const float*)d_in[19];
    const float* bv  = (const float*)d_in[20];
    float* out = (float*)d_out;

    // workspace carve-out (256B aligned regions)
    size_t off = 0;
    auto carve = [&](size_t bytes) -> char* {
        char* p = (char*)d_ws + off;
        off += (bytes + 255) & ~(size_t)255;
        return p;
    };
    const size_t MM  = (size_t)DIMS * DIMS;                 // 4M elems
    float*          WxT   = (float*)         carve((size_t)VOCABN * DIMS * 4);
    unsigned short* x_bf  = (unsigned short*)carve((size_t)TSTEPS * DIMS * 2);
    unsigned short* W4bf  = (unsigned short*)carve(4 * MM * 2);
    unsigned short* U4bf  = (unsigned short*)carve(4 * MM * 2);
    unsigned short* Vbf   = (unsigned short*)carve((size_t)VOCABN * DIMS * 2);
    float*          gates = (float*)         carve((size_t)4 * TSTEPS * DIMS * 4);
    unsigned short* hs_bf = (unsigned short*)carve((size_t)TSTEPS * DIMS * 2);
    float*          preact= (float*)         carve((size_t)2 * 8192 * 4);
    int*            barp  = (int*)           carve(256);

    (void)hipMemsetAsync(barp, 0, 256, stream);

    // 1) embedding path
    transpose_wx<<<(VOCABN * DIMS) / 256, 256, 0, stream>>>(Wx, WxT);
    embed_kernel<<<((size_t)TSTEPS * DIMS) / 256, 256, 0, stream>>>(tokens, WxT, bx, x_bf);

    // 2) bf16 conversions (W gates, U gates, V)
    const float* Ws[4] = {Wf, Wi, Wc, Wo};
    const float* Us[4] = {Uf, Ui, Uc, Uo};
    for (int g = 0; g < 4; ++g) {
        cvt_f32_to_bf16<<<4096, 256, 0, stream>>>(Ws[g], W4bf + g * MM, (long long)MM);
        cvt_f32_to_bf16<<<4096, 256, 0, stream>>>(Us[g], U4bf + g * MM, (long long)MM);
    }
    cvt_f32_to_bf16<<<2048, 256, 0, stream>>>(V, Vbf, (long long)VOCABN * DIMS);

    // 3) gate pre-activation GEMMs: gates[g] = x @ Wg^T + bwg   (WMMA bf16 + TDM)
    const float* bws[4] = {bwf, bwi, bwc, bwo};
    for (int g = 0; g < 4; ++g) {
        wmma_gemm_bf16<<<dim3(DIMS / TN, TSTEPS / TM), 256, 0, stream>>>(
            x_bf, W4bf + g * MM, bws[g], gates + (size_t)g * TSTEPS * DIMS,
            TSTEPS, DIMS, DIMS);
    }

    // 4) persistent recurrent scan
    lstm_scan<<<NB_SCAN, 256, 0, stream>>>(U4bf, buf_, bui, buc, buo,
                                           gates, preact, barp, hs_bf);

    // 5) logits = hs @ V^T + bv  (WMMA bf16 + TDM), then in-place log_softmax
    wmma_gemm_bf16<<<dim3(VOCABN / TN, TSTEPS / TM), 256, 0, stream>>>(
        hs_bf, Vbf, bv, out, TSTEPS, VOCABN, DIMS);
    log_softmax_rows<<<TSTEPS, 256, 0, stream>>>(out, VOCABN);
}
